// MetaSage_61718680044161
// MI455X (gfx1250) — matrix operand order repaired and checked by reading, not compiled, not run on hardware
//
#include <hip/hip_runtime.h>
#include <hip/hip_bf16.h>
#include <stdint.h>

// ---------------- problem constants (match reference) ----------------
#define N_PROD 100000
#define N_CUST 50000
#define HID    128
#define OUTD   64
#define E_PP   800000
#define E_PC   800000
#define E_LB   400000

// ---------------- CDNA5 WMMA / vector types ----------------
typedef __attribute__((ext_vector_type(16))) __bf16 v16bf;
typedef __attribute__((ext_vector_type(8)))  __bf16 v8bf;
typedef __attribute__((ext_vector_type(2)))  __bf16 v2bf;
typedef __attribute__((ext_vector_type(8)))  float  v8f;
typedef __attribute__((ext_vector_type(4)))  unsigned int u32x4;
typedef __attribute__((ext_vector_type(8)))  int    i32x8;
typedef __attribute__((ext_vector_type(4)))  int    i32x4;

#if __has_builtin(__builtin_amdgcn_tensor_load_to_lds) && __has_builtin(__builtin_amdgcn_s_wait_tensorcnt)
#define HAVE_TDM 1
#endif

// ---------------- fp32 -> bf16 helpers ----------------
static __device__ __forceinline__ unsigned bfbits(float f) {
    union { float f; unsigned u; } a; a.f = f;
    unsigned r = a.u + 0x7FFFu + ((a.u >> 16) & 1u);   // RNE
    return r >> 16;
}

static __device__ __forceinline__ unsigned pack_bf16x2(float x, float y) {
#if __has_builtin(__builtin_amdgcn_cvt_pk_bf16_f32)
    v2bf p = __builtin_amdgcn_cvt_pk_bf16_f32(x, y);
    union { v2bf v; unsigned u; } c; c.v = p; return c.u;
#else
    return bfbits(x) | (bfbits(y) << 16);
#endif
}

static __device__ __forceinline__ __bf16 f2bf(float f) {
    union { unsigned short s; __bf16 b; } b;
    b.s = (unsigned short)bfbits(f);
    return b.b;
}

// =====================================================================
// utility kernels: zero / convert / mean-finalize  (bandwidth-bound)
// =====================================================================
__global__ void zero_kernel(float* __restrict__ p, size_t n) {
    size_t i = (size_t)blockIdx.x * blockDim.x + threadIdx.x;
    size_t stride = (size_t)gridDim.x * blockDim.x;
    for (; i < n; i += stride) p[i] = 0.0f;
}

__global__ void cvt_bf16_kernel(const float* __restrict__ in, unsigned* __restrict__ out, size_t nPairs) {
    size_t i = (size_t)blockIdx.x * blockDim.x + threadIdx.x;
    size_t stride = (size_t)gridDim.x * blockDim.x;
    const float2* __restrict__ in2 = (const float2*)in;
    for (; i < nPairs; i += stride) {
        float2 f = in2[i];
        out[i] = pack_bf16x2(f.x, f.y);
    }
}

// mean = acc / max(cnt,1), emitted as bf16 (fused with the conversion pass)
__global__ void mean_bf16_kernel(const float* __restrict__ acc, const float* __restrict__ cnt,
                                 unsigned* __restrict__ out, size_t nPairs) {
    size_t i = (size_t)blockIdx.x * blockDim.x + threadIdx.x;
    size_t stride = (size_t)gridDim.x * blockDim.x;
    const float2* __restrict__ acc2 = (const float2*)acc;
    for (; i < nPairs; i += stride) {
        size_t row = (i * 2) >> 7;           // HID = 128
        float inv = 1.0f / fmaxf(cnt[row], 1.0f);
        float2 f = acc2[i];
        out[i] = pack_bf16x2(f.x * inv, f.y * inv);
    }
}

// =====================================================================
// degree counts + feature scatter-add (fp32 atomics, L2-resident accs)
// =====================================================================
__global__ void count_kernel(const int* __restrict__ dst, float* __restrict__ cnt, int E) {
    int e = blockIdx.x * blockDim.x + threadIdx.x;
    if (e < E) atomicAdd(&cnt[dst[e]], 1.0f);
}

__global__ __launch_bounds__(HID)
void scatter_feat_kernel(const int* __restrict__ src, const int* __restrict__ dst,
                         const float* __restrict__ x, float* __restrict__ acc) {
    int e = blockIdx.x;
    int s = src[e], d = dst[e];
    atomicAdd(&acc[(size_t)d * HID + threadIdx.x],
              x[(size_t)s * HID + threadIdx.x]);
}

// =====================================================================
// TDM staging: load a 16x128 bf16 row-major tile (4KB) from global -> LDS.
// D# built per ISA ch.8: 2D tensor, data_size=2B, tile_dim=(128,16), type=2.
// Fallback: cooperative 16B copies by one wave.
// =====================================================================
static __device__ __forceinline__ void stage_tile_16x128(__bf16* sdst, const __bf16* gsrc, int lane) {
#ifdef HAVE_TDM
    (void)lane;
    uint64_t ga = (uint64_t)(uintptr_t)gsrc;
    uint32_t lds_off = (uint32_t)(uintptr_t)sdst;      // generic shared ptr: low 32 bits = LDS offset
    u32x4 g0;
    g0[0] = 1u;                                        // count=1, user descriptor
    g0[1] = lds_off;                                   // lds_addr (bytes)
    g0[2] = (uint32_t)ga;                              // global_addr[31:0]
    g0[3] = (uint32_t)((ga >> 32) & 0x01FFFFFFu) | (2u << 30); // addr[56:32] | type=2
    i32x8 g1;
    g1[0] = (int)(1u << 16);                           // data_size=1 (2 bytes), mask=0
    g1[1] = (int)(128u << 16);                         // tensor_dim0 = 128 (low16)
    g1[2] = (int)(16u << 16);                          // tensor_dim1 = 16 (low16)
    g1[3] = (int)(128u << 16);                         // tile_dim0 = 128
    g1[4] = 16;                                        // tile_dim1 = 16, tile_dim2 = 0
    g1[5] = 128;                                       // tensor_dim0_stride = 128 (low32)
    g1[6] = 0;
    g1[7] = 0;
    i32x4 g2; g2[0] = 0; g2[1] = 0; g2[2] = 0; g2[3] = 0;
    i32x4 g3; g3[0] = 0; g3[1] = 0; g3[2] = 0; g3[3] = 0;
    i32x8 g4; g4[0] = 0; g4[1] = 0; g4[2] = 0; g4[3] = 0; g4[4] = 0; g4[5] = 0; g4[6] = 0; g4[7] = 0;
    __builtin_amdgcn_tensor_load_to_lds(g0, g1, g2, g3, g4, 0);
#else
    const uint4* __restrict__ s = (const uint4*)gsrc;  // 256 x 16B chunks
    uint4* d = (uint4*)sdst;
    #pragma unroll
    for (int i = lane; i < 256; i += 32) d[i] = s[i];
#endif
}

static __device__ __forceinline__ void stage_wait() {
#ifdef HAVE_TDM
    __builtin_amdgcn_s_wait_tensorcnt(0);
#endif
}

// A-fragment (16-bit A 16x32, wave32, ISA 7.12.2) from a row-major bf16 tile:
//   elements 0..7  -> K = kb + kh*8 + j      (16B aligned contiguous)
//   elements 8..15 -> K = kb + 16 + kh*8 + j
static __device__ __forceinline__ v16bf afrag(const __bf16* t, int m, int kb, int kh) {
    v8bf lo = *(const v8bf*)(t + m * HID + kb + kh * 8);
    v8bf hi = *(const v8bf*)(t + m * HID + kb + 16 + kh * 8);
    return __builtin_shufflevector(lo, hi, 0,1,2,3,4,5,6,7,8,9,10,11,12,13,14,15);
}

// B-fragment (32x16): lane column n; elements 0..7 -> K = kb + kh*8 + j,
// elements 8..15 -> K = kb + 16 + kh*8 + j (two 16B chunks of row n).
static __device__ __forceinline__ v16bf bfrag(const __bf16* W, int n, int kb, int kh, int ld) {
    v8bf lo = *(const v8bf*)(W + (size_t)n * ld + kb + kh * 8);
    v8bf hi = *(const v8bf*)(W + (size_t)n * ld + kb + 16 + kh * 8);
    return __builtin_shufflevector(lo, hi, 0,1,2,3,4,5,6,7,8,9,10,11,12,13,14,15);
}

// =====================================================================
// SAGE layer: out = act( mean @ Wl^T + bl + xdst @ Wr^T )
// block = 128 thr = 4 waves; 16 rows x 128 cols; wave w -> col tiles {2w,2w+1}.
// A tiles staged via TDM into LDS (shared by all 4 waves); pure load+WMMA loop.
// =====================================================================
__global__ __launch_bounds__(128)
void sage_layer_kernel(const __bf16* __restrict__ meanb, const __bf16* __restrict__ xdstb,
                       const __bf16* __restrict__ Wl, const float* __restrict__ bl,
                       const __bf16* __restrict__ Wr,
                       float* __restrict__ outF, __bf16* __restrict__ outH, int doRelu) {
    __shared__ __bf16 sM[16 * HID];
    __shared__ __bf16 sX[16 * HID];
    const int lane = threadIdx.x & 31;
    const int wave = threadIdx.x >> 5;
    const int m    = lane & 15;
    const int kh   = lane >> 4;
    const size_t rowbase = (size_t)blockIdx.x * 16;

    if (wave == 0) {
        stage_tile_16x128(sM, meanb + rowbase * HID, lane);
        stage_tile_16x128(sX, xdstb + rowbase * HID, lane);
        stage_wait();
    }
    __syncthreads();

    v8f c0 = {}; v8f c1 = {};
    #pragma unroll
    for (int ks = 0; ks < 4; ++ks) {
        const int kb = ks * 32;
        const v16bf aM = afrag(sM, m, kb, kh);
        const v16bf aX = afrag(sX, m, kb, kh);

        const int n0 = wave * 32 + m;
        const v16bf bL0 = bfrag(Wl, n0,      kb, kh, HID);
        const v16bf bR0 = bfrag(Wr, n0,      kb, kh, HID);
        const v16bf bL1 = bfrag(Wl, n0 + 16, kb, kh, HID);
        const v16bf bR1 = bfrag(Wr, n0 + 16, kb, kh, HID);

        c0 = __builtin_amdgcn_wmma_f32_16x16x32_bf16(false, aM, false, bL0, (short)0, c0, false, false);
        c0 = __builtin_amdgcn_wmma_f32_16x16x32_bf16(false, aX, false, bR0, (short)0, c0, false, false);
        c1 = __builtin_amdgcn_wmma_f32_16x16x32_bf16(false, aM, false, bL1, (short)0, c1, false, false);
        c1 = __builtin_amdgcn_wmma_f32_16x16x32_bf16(false, aX, false, bR1, (short)0, c1, false, false);
    }

    // D layout: element v -> row (v + 8*kh), col m of the tile
    #pragma unroll
    for (int t = 0; t < 2; ++t) {
        const int n = wave * 32 + t * 16 + m;
        const float bias = bl[n];
        const v8f cc = t ? c1 : c0;
        #pragma unroll
        for (int v = 0; v < 8; ++v) {
            const size_t orow = rowbase + v + kh * 8;
            float val = cc[v] + bias;
            if (doRelu) val = fmaxf(val, 0.0f);
            if (outF) outF[orow * HID + n] = val;
            if (outH) outH[orow * HID + n] = f2bf(val);
        }
    }
}

// =====================================================================
// Linear 128 -> 64 (bf16 in, bf16 out): out = x @ W^T + b
// block = 128 thr = 4 waves; 16 rows x 64 cols; wave w -> col tile w.
// x tile staged via TDM into LDS.
// =====================================================================
__global__ __launch_bounds__(128)
void linear_out_kernel(const __bf16* __restrict__ x, const __bf16* __restrict__ W,
                       const float* __restrict__ b, __bf16* __restrict__ outH) {
    __shared__ __bf16 sA[16 * HID];
    const int lane = threadIdx.x & 31;
    const int wave = threadIdx.x >> 5;
    const int m    = lane & 15;
    const int kh   = lane >> 4;
    const size_t rowbase = (size_t)blockIdx.x * 16;

    if (wave == 0) {
        stage_tile_16x128(sA, x + rowbase * HID, lane);
        stage_wait();
    }
    __syncthreads();

    v8f c = {};
    const int n = wave * 16 + m;
    #pragma unroll
    for (int ks = 0; ks < 4; ++ks) {
        const int kb = ks * 32;
        const v16bf a  = afrag(sA, m, kb, kh);
        const v16bf bb = bfrag(W, n, kb, kh, HID);
        c = __builtin_amdgcn_wmma_f32_16x16x32_bf16(false, a, false, bb, (short)0, c, false, false);
    }

    const float bias = b[n];
    #pragma unroll
    for (int v = 0; v < 8; ++v) {
        const size_t orow = rowbase + v + kh * 8;
        outH[orow * OUTD + n] = f2bf(c[v] + bias);
    }
}

// =====================================================================
// Fused edge decoder:
//   h = relu([z_cust[row], z_prod[col]] @ W1^T + b1);  out = h.w2 + b2
// Gathered bf16 rows feed the A-fragments directly (concat is free);
// the 64->1 dot is reduced in LDS via ds_add_f32.
// =====================================================================
__global__ __launch_bounds__(128)
void decoder_kernel(const __bf16* __restrict__ z_cust, const __bf16* __restrict__ z_prod,
                    const int* __restrict__ rowi, const int* __restrict__ coli,
                    const __bf16* __restrict__ W1, const float* __restrict__ b1,
                    const float* __restrict__ w2, const float* __restrict__ b2,
                    float* __restrict__ out) {
    __shared__ float red[16];
    const int lane = threadIdx.x & 31;
    const int wave = threadIdx.x >> 5;
    const int m    = lane & 15;
    const int kh   = lane >> 4;
    const int e    = blockIdx.x * 16 + m;

    const __bf16* __restrict__ zc = z_cust + (size_t)rowi[e] * OUTD;
    const __bf16* __restrict__ zp = z_prod + (size_t)coli[e] * OUTD;

    if (threadIdx.x < 16) red[threadIdx.x] = 0.0f;
    __syncthreads();

    v8f c = {};
    const int n = wave * 16 + m;
    #pragma unroll
    for (int ks = 0; ks < 4; ++ks) {
        const __bf16* __restrict__ srcp = (ks < 2) ? zc : zp;
        const int kb = (ks & 1) * 32;
        v8bf lo = *(const v8bf*)(srcp + kb + kh * 8);
        v8bf hi = *(const v8bf*)(srcp + kb + 16 + kh * 8);
        v16bf a = __builtin_shufflevector(lo, hi, 0,1,2,3,4,5,6,7,8,9,10,11,12,13,14,15);
        v16bf bb = bfrag(W1, n, ks * 32, kh, 2 * OUTD);
        c = __builtin_amdgcn_wmma_f32_16x16x32_bf16(false, a, false, bb, (short)0, c, false, false);
    }

    const float bias = b1[n];
    const float wn   = w2[n];
    #pragma unroll
    for (int v = 0; v < 8; ++v) {
        const float h = fmaxf(c[v] + bias, 0.0f);
        atomicAdd(&red[v + kh * 8], h * wn);     // ds_add_f32
    }
    __syncthreads();
    if (threadIdx.x < 16)
        out[(size_t)blockIdx.x * 16 + threadIdx.x] = red[threadIdx.x] + b2[0];
}

// =====================================================================
// launch
// =====================================================================
extern "C" void kernel_launch(void* const* d_in, const int* in_sizes, int n_in,
                              void* d_out, int out_size, void* d_ws, size_t ws_size,
                              hipStream_t stream) {
    const float* x_prod  = (const float*)d_in[0];
    const float* x_cust  = (const float*)d_in[1];
    const int*   ei_pp   = (const int*)d_in[2];
    const int*   ei_pc   = (const int*)d_in[3];
    const int*   eli     = (const int*)d_in[4];
    const float* it_W1l  = (const float*)d_in[5];
    const float* it_W1r  = (const float*)d_in[6];
    const float* it_b1   = (const float*)d_in[7];
    const float* it_W2l  = (const float*)d_in[8];
    const float* it_W2r  = (const float*)d_in[9];
    const float* it_b2   = (const float*)d_in[10];
    const float* it_Wlin = (const float*)d_in[11];
    const float* it_blin = (const float*)d_in[12];
    const float* us_W1l  = (const float*)d_in[13];
    const float* us_W1r  = (const float*)d_in[14];
    const float* us_b1   = (const float*)d_in[15];
    const float* us_W2l  = (const float*)d_in[16];
    const float* us_W2r  = (const float*)d_in[17];
    const float* us_b2   = (const float*)d_in[18];
    const float* us_W3l  = (const float*)d_in[19];
    const float* us_W3r  = (const float*)d_in[20];
    const float* us_b3   = (const float*)d_in[21];
    const float* us_Wlin = (const float*)d_in[22];
    const float* us_blin = (const float*)d_in[23];
    const float* de_W1   = (const float*)d_in[24];
    const float* de_b1   = (const float*)d_in[25];
    const float* de_W2   = (const float*)d_in[26];
    const float* de_b2   = (const float*)d_in[27];

    const int* pp_s = ei_pp;  const int* pp_d = ei_pp + E_PP;
    const int* pc_s = ei_pc;  const int* pc_d = ei_pc + E_PC;
    const int* lb_r = eli;    const int* lb_c = eli + E_LB;

    // ---------------- workspace layout ----------------
    float* ws = (float*)d_ws;
    size_t o = 0;
    float* acc_p = ws + o; o += (size_t)N_PROD * HID;
    float* acc_c = ws + o; o += (size_t)N_CUST * HID;
    float* cnt_p = ws + o; o += N_PROD;
    float* cnt_c = ws + o; o += N_CUST;
    float* p1F   = ws + o; o += (size_t)N_PROD * HID;   // scatter input (fp32)
    float* pxF   = ws + o; o += (size_t)N_PROD * HID;   // scatter input (fp32)

    __bf16* hb = (__bf16*)(ws + o);
    size_t h = 0;
    __bf16* meanB = hb + h; h += (size_t)N_PROD * HID;  // shared mean buffer (prod- or cust-sized)
    __bf16* xpB   = hb + h; h += (size_t)N_PROD * HID;
    __bf16* xcB   = hb + h; h += (size_t)N_CUST * HID;
    __bf16* p1B   = hb + h; h += (size_t)N_PROD * HID;
    __bf16* cxB   = hb + h; h += (size_t)N_CUST * HID;
    __bf16* p2B   = hb + h; h += (size_t)N_PROD * HID;
    __bf16* cx2B  = hb + h; h += (size_t)N_CUST * HID;
    __bf16* zpB   = hb + h; h += (size_t)N_PROD * OUTD;
    __bf16* zcB   = hb + h; h += (size_t)N_CUST * OUTD;
    // bf16 weights
    __bf16* wIt1l = hb + h; h += HID * HID;
    __bf16* wIt1r = hb + h; h += HID * HID;
    __bf16* wIt2l = hb + h; h += HID * HID;
    __bf16* wIt2r = hb + h; h += HID * HID;
    __bf16* wUs1l = hb + h; h += HID * HID;
    __bf16* wUs1r = hb + h; h += HID * HID;
    __bf16* wUs2l = hb + h; h += HID * HID;
    __bf16* wUs2r = hb + h; h += HID * HID;
    __bf16* wUs3l = hb + h; h += HID * HID;
    __bf16* wUs3r = hb + h; h += HID * HID;
    __bf16* wItL  = hb + h; h += OUTD * HID;
    __bf16* wUsL  = hb + h; h += OUTD * HID;
    __bf16* wDe1  = hb + h; h += OUTD * (2 * OUTD);

    float* outp = (float*)d_out;

    const dim3 B128(128), B256(256);
    const dim3 gZero(2048), gCvt(2048);
    const dim3 gProd(N_PROD / 16), gCust(N_CUST / 16), gLab(E_LB / 16);

    // ---- pre-convert weights + node features to bf16 (one pass) ----
    {
        struct { const float* s; __bf16* d; size_t n; } cv[] = {
            { x_prod, xpB, (size_t)N_PROD * HID }, { x_cust, xcB, (size_t)N_CUST * HID },
            { it_W1l, wIt1l, HID * HID }, { it_W1r, wIt1r, HID * HID },
            { it_W2l, wIt2l, HID * HID }, { it_W2r, wIt2r, HID * HID },
            { us_W1l, wUs1l, HID * HID }, { us_W1r, wUs1r, HID * HID },
            { us_W2l, wUs2l, HID * HID }, { us_W2r, wUs2r, HID * HID },
            { us_W3l, wUs3l, HID * HID }, { us_W3r, wUs3r, HID * HID },
            { it_Wlin, wItL, OUTD * HID }, { us_Wlin, wUsL, OUTD * HID },
            { de_W1, wDe1, OUTD * 2 * OUTD },
        };
        for (auto& c : cv)
            cvt_bf16_kernel<<<gCvt, B256, 0, stream>>>(c.s, (unsigned*)c.d, c.n / 2);
    }

    // ---- degree counts (structure-only) ----
    zero_kernel<<<gZero, B256, 0, stream>>>(cnt_p, (size_t)N_PROD);
    zero_kernel<<<gZero, B256, 0, stream>>>(cnt_c, (size_t)N_CUST);
    count_kernel<<<(E_PP + 255) / 256, B256, 0, stream>>>(pp_d, cnt_p, E_PP);
    count_kernel<<<(E_PC + 255) / 256, B256, 0, stream>>>(pc_d, cnt_c, E_PC);

    // ---- pp aggregation of x_product (shared by item-L1 and user-L1) ----
    zero_kernel<<<gZero, B256, 0, stream>>>(acc_p, (size_t)N_PROD * HID);
    scatter_feat_kernel<<<E_PP, B128, 0, stream>>>(pp_s, pp_d, x_prod, acc_p);
    mean_bf16_kernel<<<gCvt, B256, 0, stream>>>(acc_p, cnt_p, (unsigned*)meanB, (size_t)N_PROD * HID / 2);
    sage_layer_kernel<<<gProd, B128, 0, stream>>>(meanB, xpB, wIt1l, it_b1, wIt1r, p1F, p1B, 1);
    sage_layer_kernel<<<gProd, B128, 0, stream>>>(meanB, xpB, wUs1l, us_b1, wUs1r, pxF, nullptr, 1);

    // ---- user-L2: pc aggregation of x_product into customers ----
    zero_kernel<<<gZero, B256, 0, stream>>>(acc_c, (size_t)N_CUST * HID);
    scatter_feat_kernel<<<E_PC, B128, 0, stream>>>(pc_s, pc_d, x_prod, acc_c);
    mean_bf16_kernel<<<gCvt, B256, 0, stream>>>(acc_c, cnt_c, (unsigned*)meanB, (size_t)N_CUST * HID / 2);
    sage_layer_kernel<<<gCust, B128, 0, stream>>>(meanB, xcB, wUs2l, us_b2, wUs2r, nullptr, cxB, 1);

    // ---- item-L2 on p1, then item projection ----
    zero_kernel<<<gZero, B256, 0, stream>>>(acc_p, (size_t)N_PROD * HID);
    scatter_feat_kernel<<<E_PP, B128, 0, stream>>>(pp_s, pp_d, p1F, acc_p);
    mean_bf16_kernel<<<gCvt, B256, 0, stream>>>(acc_p, cnt_p, (unsigned*)meanB, (size_t)N_PROD * HID / 2);
    sage_layer_kernel<<<gProd, B128, 0, stream>>>(meanB, p1B, wIt2l, it_b2, wIt2r, nullptr, p2B, 1);
    linear_out_kernel<<<gProd, B128, 0, stream>>>(p2B, wItL, it_blin, zpB);

    // ---- user-L3: pc aggregation of px into customers, then user projection ----
    zero_kernel<<<gZero, B256, 0, stream>>>(acc_c, (size_t)N_CUST * HID);
    scatter_feat_kernel<<<E_PC, B128, 0, stream>>>(pc_s, pc_d, pxF, acc_c);
    mean_bf16_kernel<<<gCvt, B256, 0, stream>>>(acc_c, cnt_c, (unsigned*)meanB, (size_t)N_CUST * HID / 2);
    sage_layer_kernel<<<gCust, B128, 0, stream>>>(meanB, cxB, wUs3l, us_b3, wUs3r, nullptr, cx2B, 1);
    linear_out_kernel<<<gCust, B128, 0, stream>>>(cx2B, wUsL, us_blin, zcB);

    // ---- fused edge decoder ----
    decoder_kernel<<<gLab, B128, 0, stream>>>(zcB, zpB, lb_r, lb_c,
                                              wDe1, de_b1, de_W2, de_b2, outp);
}
